// RSSM_18416819765254
// MI455X (gfx1250) — matrix operand order repaired and checked by reading, not compiled, not run on hardware
//
#include <hip/hip_runtime.h>
#include <hip/hip_bf16.h>
#include <math.h>

// ============================================================================
// RSSM posterior scan for gfx1250 (MI455X).
// - recurrence independent per batch row -> each wave32 owns a 16-row tile
//   (WMMA M dim) and runs all 256 timesteps with activations in LDS.
// - matmuls: v_wmma_f32_16x16x32_f16 (f32 accumulate), weights converted once
//   to f16 in workspace, streamed from L2/WGP$.
// - nonlinearities: native v_tanh_f32 (sigmoid via 0.5*tanh(x/2)+0.5) so the
//   TRANS pipe co-executes with the WMMA pipe instead of branchy libm tanh.
// ============================================================================

typedef __attribute__((ext_vector_type(16))) _Float16 v16h;
typedef __attribute__((ext_vector_type(4)))  _Float16 v4h;
typedef __attribute__((ext_vector_type(8)))  float    v8f;

static constexpr int L_SEQ = 256;
static constexpr int BATCH = 1024;
static constexpr int WAVES = 2;          // waves per workgroup (independent tiles)

// ---- f16 weight offsets inside workspace (units: halves) -------------------
static constexpr int W_U1  = 0;          // 64x32
static constexpr int W_U2  = 2048;       // 64x64
static constexpr int W_X1  = 6144;       // 64x64
static constexpr int W_X2  = 10240;      // 64x64
static constexpr int W_S1  = 14336;      // 64x32
static constexpr int W_S2  = 16384;      // 64x64
static constexpr int W_HX1 = 20480;      // 64x128
static constexpr int W_HX2 = 28672;      // 64x64
static constexpr int W_SA1 = 32768;      // 64x128
static constexpr int W_SA2 = 40960;      // 64x64
static constexpr int W_P1  = 45056;      // 128x64
static constexpr int W_P2  = 53248;      // 128x64
static constexpr int W_PMU = 61440;      // 32x128
static constexpr int W_PLS = 65536;      // 32x128
static constexpr int W_GIH = 69632;      // 192x64
static constexpr int W_GHH = 81920;      // 192x64

union H16 { v16h v; unsigned u[8]; };

// ---- native transcendentals (gfx1250 V_TANH_F32 / V_EXP_F32) ---------------
__device__ __forceinline__ float fast_tanh(float x) {
#if __has_builtin(__builtin_amdgcn_tanhf)
    return __builtin_amdgcn_tanhf(x);
#elif __has_builtin(__builtin_amdgcn_tanh_f32)
    return __builtin_amdgcn_tanh_f32(x);
#else
    // branchless fallback: tanh(x) = 1 - 2/(exp2(2x*log2e)+1)
    float e = __builtin_amdgcn_exp2f(x * 2.8853900817779268f);
    return 1.0f - 2.0f * __builtin_amdgcn_rcpf(e + 1.0f);
#endif
}
__device__ __forceinline__ float fast_sigm(float x) {
    return 0.5f * fast_tanh(0.5f * x) + 0.5f;
}
__device__ __forceinline__ float fast_exp(float x) {
#if __has_builtin(__builtin_amdgcn_exp2f)
    return __builtin_amdgcn_exp2f(x * 1.4426950408889634f);
#else
    return __expf(x);
#endif
}

// A-tile: 16 rows x 32 K from row-major f16 LDS (stride halves).
__device__ __forceinline__ v16h load_A(const _Float16* src, int stride, int k0, int lane) {
    int m  = lane & 15;
    int kb = (lane >> 4) * 8;
    const _Float16* p = src + m * stride + k0;
    H16 r;
#pragma unroll
    for (int v = 0; v < 8; ++v) {
        int k = ((v & 4) ? 16 : 0) + kb + 2 * (v & 3);
        r.u[v] = *(const unsigned*)(p + k);
    }
    return r.v;
}

// B-tile: 32 K x 16 N from f16 weights stored [out][in] row-major.
__device__ __forceinline__ v16h load_B(const _Float16* w, int in_dim, int n0, int k0, int lane) {
    int n  = lane & 15;
    int kb = (lane >> 4) * 8;
    const _Float16* p = w + (size_t)(n0 + n) * in_dim + k0 + kb;
    H16 r;
#pragma unroll
    for (int v = 0; v < 8; ++v) r.u[v] = *(const unsigned*)(p + ((v & 4) ? 16 : 0) + 2 * (v & 3));
    return r.v;
}

template <int IN>
__device__ __forceinline__ v8f mm_tile(const _Float16* src, const _Float16* w,
                                       int n0, int lane, v8f acc) {
#pragma unroll
    for (int k0 = 0; k0 < IN; k0 += 32) {
        v16h a = load_A(src, IN, k0, lane);
        v16h b = load_B(w, IN, n0, k0, lane);
        acc = __builtin_amdgcn_wmma_f32_16x16x32_f16(false, a, false, b,
                                                     (short)0, acc, false, false);
    }
    return acc;
}

// Linear layer + activation, f16 result into LDS. ACT: 0=tanh, 1=sigmoid, 2=id
template <int IN, int OUT, int ACT>
__device__ __forceinline__ void layer(const _Float16* src, const _Float16* w,
                                      const float* bias, _Float16* dst,
                                      int dstStride, int dstCol0, int lane) {
    int n  = lane & 15;
    int mb = (lane >> 4) * 8;
#pragma unroll
    for (int nt = 0; nt < OUT / 16; ++nt) {
        v8f acc = {};
        acc = mm_tile<IN>(src, w, nt * 16, lane, acc);
        float b = bias[nt * 16 + n];
#pragma unroll
        for (int v = 0; v < 8; ++v) {
            float x = acc[v] + b;
            if (ACT == 0) x = fast_tanh(x);
            else if (ACT == 1) x = fast_sigm(x);
            dst[(mb + v) * dstStride + dstCol0 + nt * 16 + n] = (_Float16)x;
        }
    }
}

struct ScanArgs {
    const _Float16* w;
    const float *bu1, *bu2, *bx1, *bx2, *bs1, *bs2;
    const float *bhx1, *bhx2, *bsa1, *bsa2;
    const float *bp1, *bp2, *bmu, *bls;
    const float *bih, *bhh;
    const float *u, *x, *noise;
    float *out_mu, *out_ls, *out_s, *out_h;
};

__global__ __launch_bounds__(32 * WAVES) void rssm_scan(ScanArgs A) {
    __shared__ alignas(16) float    hnF [WAVES][16 * 64];   // hidden state (f32)
    __shared__ alignas(16) _Float16 cat [WAVES][16 * 128];  // concat buffer
    __shared__ alignas(16) _Float16 tbuf[WAVES][16 * 128];  // dblock gated t
    __shared__ alignas(16) _Float16 b0  [WAVES][16 * 64];
    __shared__ alignas(16) _Float16 b1  [WAVES][16 * 64];
    __shared__ alignas(16) _Float16 ueb [WAVES][16 * 64];   // u embedding
    __shared__ alignas(16) _Float16 s16 [WAVES][16 * 32];   // sampled state f16
    __shared__ alignas(16) float    nzF [WAVES][16 * 32];   // staged noise
    __shared__ alignas(16) float    muF [WAVES][16 * 32];   // staged mu
    __shared__ alignas(16) float    lsF [WAVES][16 * 32];   // staged logsigma
    __shared__ alignas(16) float    ssF [WAVES][16 * 32];   // staged s

    const int lane    = threadIdx.x & 31;
    const int wv      = threadIdx.x >> 5;
    const int rowBase = (blockIdx.x * WAVES + wv) * 16;

    float*    hn = hnF[wv];
    _Float16 *CAT = cat[wv], *T = tbuf[wv], *B0 = b0[wv], *B1 = b1[wv];
    _Float16 *UE = ueb[wv], *S = s16[wv];
    float *NZ = nzF[wv], *MU = muF[wv], *LS = lsF[wv], *SS = ssF[wv];

    for (int i = lane; i < 16 * 64; i += 32) hn[i] = 0.0f;

    const _Float16* W = A.w;
    const int n  = lane & 15;
    const int mb = (lane >> 4) * 8;

    for (int l = 0; l < L_SEQ; ++l) {
        if (l + 1 < L_SEQ) {   // global_prefetch_b8 of next step's inputs
            __builtin_prefetch(A.x + (size_t)(l + 1) * BATCH * 64 + (size_t)rowBase * 64 + lane * 32, 0, 1);
            __builtin_prefetch(A.u + (size_t)(l + 1) * BATCH * 32 + (size_t)rowBase * 32 + lane * 16, 0, 1);
            __builtin_prefetch(A.noise + (size_t)(l + 1) * BATCH * 32 + (size_t)rowBase * 32 + lane * 16, 0, 1);
        }

        // ---- stage inputs (vectorized b128 loads) --------------------------
        const float4* usrc = (const float4*)(A.u + (size_t)l * BATCH * 32 + (size_t)rowBase * 32);
        for (int i = lane; i < 16 * 32 / 4; i += 32) {
            float4 f = usrc[i];
            v4h h4 = { (_Float16)f.x, (_Float16)f.y, (_Float16)f.z, (_Float16)f.w };
            *(v4h*)(B0 + i * 4) = h4;
        }
        const float4* nsrc = (const float4*)(A.noise + (size_t)l * BATCH * 32 + (size_t)rowBase * 32);
        for (int i = lane; i < 16 * 32 / 4; i += 32) ((float4*)NZ)[i] = nsrc[i];

        // ---- u_e = preprocess(u, process_u) --------------------------------
        layer<32, 64, 0>(B0, W + W_U1, A.bu1, B1, 64, 0, lane);
        layer<64, 64, 0>(B1, W + W_U2, A.bu2, UE, 64, 0, lane);

        // ---- x_e = preprocess(x, process_x) -> CAT[:,64:128] ---------------
        const float4* xsrc = (const float4*)(A.x + (size_t)l * BATCH * 64 + (size_t)rowBase * 64);
        for (int i = lane; i < 16 * 64 / 4; i += 32) {
            float4 f = xsrc[i];
            v4h h4 = { (_Float16)f.x, (_Float16)f.y, (_Float16)f.z, (_Float16)f.w };
            *(v4h*)(B0 + i * 4) = h4;
        }
        layer<64, 64, 0>(B0, W + W_X1, A.bx1, B1, 64, 0, lane);
        layer<64, 64, 0>(B1, W + W_X2, A.bx2, CAT, 128, 64, lane);

        // hn (f32) -> CAT[:,0:64] as f16 (b128 LDS reads, b64 writes)
        for (int i = lane; i < 16 * 64 / 4; i += 32) {
            float4 f = ((const float4*)hn)[i];
            v4h h4 = { (_Float16)f.x, (_Float16)f.y, (_Float16)f.z, (_Float16)f.w };
            int row = i >> 4, c4 = i & 15;
            *(v4h*)(CAT + row * 128 + c4 * 4) = h4;
        }

        // ---- hidden = preprocess([hn | x_e], process_hx) -------------------
        layer<128, 64, 0>(CAT, W + W_HX1, A.bhx1, B0, 64, 0, lane);
        layer<64,  64, 0>(B0,  W + W_HX2, A.bhx2, B1, 64, 0, lane);

        // ---- dblock gate: t = tanh(w1 h) * sigmoid(w2 h) -> T (16x128) -----
#pragma unroll
        for (int nt = 0; nt < 8; ++nt) {
            v8f a1 = {}; a1 = mm_tile<64>(B1, W + W_P1, nt * 16, lane, a1);
            v8f a2 = {}; a2 = mm_tile<64>(B1, W + W_P2, nt * 16, lane, a2);
            float c1 = A.bp1[nt * 16 + n], c2 = A.bp2[nt * 16 + n];
#pragma unroll
            for (int v = 0; v < 8; ++v) {
                float t = fast_tanh(a1[v] + c1) * fast_sigm(a2[v] + c2);
                T[(mb + v) * 128 + nt * 16 + n] = (_Float16)t;
            }
        }

        // ---- mu, logsigma, s = mu + exp(ls)*eps ----------------------------
#pragma unroll
        for (int nt = 0; nt < 2; ++nt) {
            v8f am = {}; am = mm_tile<128>(T, W + W_PMU, nt * 16, lane, am);
            v8f al = {}; al = mm_tile<128>(T, W + W_PLS, nt * 16, lane, al);
            float cm = A.bmu[nt * 16 + n], cl = A.bls[nt * 16 + n];
#pragma unroll
            for (int v = 0; v < 8; ++v) {
                int m = mb + v, col = nt * 16 + n;
                float mu = am[v] + cm;
                float ls = al[v] + cl;
                float s  = mu + fast_exp(ls) * NZ[m * 32 + col];
                MU[m * 32 + col] = mu;
                LS[m * 32 + col] = ls;
                SS[m * 32 + col] = s;
                S [m * 32 + col] = (_Float16)s;
            }
        }
        // coalesced b128 stores of mu / logsigma / s
        {
            float4* omu = (float4*)(A.out_mu + (size_t)l * BATCH * 32 + (size_t)rowBase * 32);
            float4* ols = (float4*)(A.out_ls + (size_t)l * BATCH * 32 + (size_t)rowBase * 32);
            float4* oss = (float4*)(A.out_s  + (size_t)l * BATCH * 32 + (size_t)rowBase * 32);
            for (int i = lane; i < 16 * 32 / 4; i += 32) {
                omu[i] = ((const float4*)MU)[i];
                ols[i] = ((const float4*)LS)[i];
                oss[i] = ((const float4*)SS)[i];
            }
        }

        // ---- sp = preprocess(s, process_s) -> CAT[:,0:64] ------------------
        layer<32, 64, 0>(S,  W + W_S1, A.bs1, B0, 64, 0, lane);
        layer<64, 64, 0>(B0, W + W_S2, A.bs2, CAT, 128, 0, lane);
        // u_e -> CAT[:,64:128] (b64 LDS-to-LDS copies)
        for (int i = lane; i < 16 * 64 / 4; i += 32) {
            int row = i >> 4, c4 = i & 15;
            *(v4h*)(CAT + row * 128 + 64 + c4 * 4) = *(const v4h*)(UE + row * 64 + c4 * 4);
        }

        // ---- h2 = preprocess([sp | u_e], process_sa) -----------------------
        layer<128, 64, 0>(CAT, W + W_SA1, A.bsa1, B0, 64, 0, lane);
        layer<64,  64, 0>(B0,  W + W_SA2, A.bsa2, B1, 64, 0, lane);

        // ---- emit pre-update hn (h_seq, b128 stores), f16 copy for GRU -----
        {
            float4* oh = (float4*)(A.out_h + (size_t)l * BATCH * 64 + (size_t)rowBase * 64);
            for (int i = lane; i < 16 * 64 / 4; i += 32) {
                float4 f = ((const float4*)hn)[i];
                oh[i] = f;
                v4h h4 = { (_Float16)f.x, (_Float16)f.y, (_Float16)f.z, (_Float16)f.w };
                *(v4h*)(B0 + i * 4) = h4;
            }
        }

        // ---- GRU: hn = (1-z)*g + z*hn, gates tile-by-tile in registers -----
#pragma unroll
        for (int nt = 0; nt < 4; ++nt) {
            v8f ir = {}, iz = {}, ig = {}, hr = {}, hz = {}, hg = {};
            ir = mm_tile<64>(B1, W + W_GIH,        nt * 16, lane, ir);
            iz = mm_tile<64>(B1, W + W_GIH,  64 +  nt * 16, lane, iz);
            ig = mm_tile<64>(B1, W + W_GIH, 128 +  nt * 16, lane, ig);
            hr = mm_tile<64>(B0, W + W_GHH,        nt * 16, lane, hr);
            hz = mm_tile<64>(B0, W + W_GHH,  64 +  nt * 16, lane, hz);
            hg = mm_tile<64>(B0, W + W_GHH, 128 +  nt * 16, lane, hg);
            float bir = A.bih[nt * 16 + n], biz = A.bih[64 + nt * 16 + n], big = A.bih[128 + nt * 16 + n];
            float bhr = A.bhh[nt * 16 + n], bhz = A.bhh[64 + nt * 16 + n], bhg = A.bhh[128 + nt * 16 + n];
#pragma unroll
            for (int v = 0; v < 8; ++v) {
                int m = mb + v, col = nt * 16 + n;
                float r = fast_sigm(ir[v] + bir + hr[v] + bhr);
                float z = fast_sigm(iz[v] + biz + hz[v] + bhz);
                float g = fast_tanh(ig[v] + big + r * (hg[v] + bhg));
                float h = hn[m * 64 + col];
                hn[m * 64 + col] = (1.0f - z) * g + z * h;
            }
        }
    }
}

__global__ void cvt_f32_f16(const float* __restrict__ s, _Float16* __restrict__ d, int nElem) {
    int i = blockIdx.x * 256 + threadIdx.x;
    if (i < nElem) d[i] = (_Float16)s[i];
}

extern "C" void kernel_launch(void* const* d_in, const int* in_sizes, int n_in,
                              void* d_out, int out_size, void* d_ws, size_t ws_size,
                              hipStream_t stream) {
    (void)in_sizes; (void)n_in; (void)out_size; (void)ws_size;

    _Float16* wbase = (_Float16*)d_ws;
    auto cvt = [&](int idx, int off, int cnt) {
        cvt_f32_f16<<<dim3((cnt + 255) / 256), dim3(256), 0, stream>>>(
            (const float*)d_in[idx], wbase + off, cnt);
    };
    // d_in order follows setup_inputs() dict flattening.
    cvt(0,  W_U1,  64 * 32);   cvt(2,  W_U2,  64 * 64);
    cvt(4,  W_X1,  64 * 64);   cvt(6,  W_X2,  64 * 64);
    cvt(8,  W_S1,  64 * 32);   cvt(10, W_S2,  64 * 64);
    cvt(12, W_HX1, 64 * 128);  cvt(14, W_HX2, 64 * 64);
    cvt(16, W_SA1, 64 * 128);  cvt(18, W_SA2, 64 * 64);
    cvt(20, W_P1,  128 * 64);  cvt(22, W_P2,  128 * 64);
    cvt(24, W_PMU, 32 * 128);  cvt(26, W_PLS, 32 * 128);
    cvt(28, W_GIH, 192 * 64);  cvt(29, W_GHH, 192 * 64);

    ScanArgs a;
    a.w    = wbase;
    a.bu1  = (const float*)d_in[1];   a.bu2  = (const float*)d_in[3];
    a.bx1  = (const float*)d_in[5];   a.bx2  = (const float*)d_in[7];
    a.bs1  = (const float*)d_in[9];   a.bs2  = (const float*)d_in[11];
    a.bhx1 = (const float*)d_in[13];  a.bhx2 = (const float*)d_in[15];
    a.bsa1 = (const float*)d_in[17];  a.bsa2 = (const float*)d_in[19];
    a.bp1  = (const float*)d_in[21];  a.bp2  = (const float*)d_in[23];
    a.bmu  = (const float*)d_in[25];  a.bls  = (const float*)d_in[27];
    a.bih  = (const float*)d_in[30];  a.bhh  = (const float*)d_in[31];
    a.u     = (const float*)d_in[32];
    a.x     = (const float*)d_in[33];
    a.noise = (const float*)d_in[34];

    float* out = (float*)d_out;
    const size_t seg = (size_t)L_SEQ * BATCH * 32;
    a.out_mu = out;
    a.out_ls = out + seg;
    a.out_s  = out + 2 * seg;
    a.out_h  = out + 3 * seg;

    rssm_scan<<<dim3(BATCH / (16 * WAVES)), dim3(32 * WAVES), 0, stream>>>(a);
}
